// MultiHeadAttention_22351009808473
// MI455X (gfx1250) — matrix operand order repaired and checked by reading, not compiled, not run on hardware
//
#include <hip/hip_runtime.h>
#include <hip/hip_bf16.h>
#include <stdint.h>

#define DIM 1024
#define SEQ 2048
#define NB  4
#define NH  16
#define DH  64

typedef __attribute__((ext_vector_type(16))) __bf16 v16bf;
typedef __attribute__((ext_vector_type(8)))  float  v8f;
typedef __attribute__((ext_vector_type(4)))  unsigned int v4u;
typedef __attribute__((ext_vector_type(4)))  int v4i;
typedef __attribute__((ext_vector_type(8)))  int v8i;

union Frag16 { v16bf v; uint4 q[2]; };

__device__ __forceinline__ unsigned short f2bf(float f) {
  unsigned int u = __float_as_uint(f);
  u += 0x7FFFu + ((u >> 16) & 1u);   // round-to-nearest-even
  return (unsigned short)(u >> 16);
}

__device__ __forceinline__ v8f wmma_bf16(const v16bf& a, const v16bf& b, const v8f& c) {
  return __builtin_amdgcn_wmma_f32_16x16x32_bf16(false, a, false, b, (short)0, c, false, false);
}

// ---- Tensor Data Mover: DMA a 2D bf16 tile (tile_d0 x tile_d1 elems) from
// global (row stride = stride0 elems) into LDS at lds_off, inserting
// pad_amt DWORDs of padding every pad_int DWORDs (codes per D# spec).
// Issued once per wave (EXEC-independent); tracked by TENSORcnt.
__device__ __forceinline__ void tdm_load_2d(unsigned lds_off,
                                            const unsigned short* gptr,
                                            unsigned tensor_d0, unsigned tensor_d1,
                                            unsigned tile_d0,   unsigned tile_d1,
                                            unsigned stride0,
                                            unsigned pad_int_code, unsigned pad_amt_code)
{
  unsigned long long ga = (unsigned long long)(uintptr_t)gptr;
  v4u g0;
  g0.x = 1u;                                   // count=1, user-mode, no gather
  g0.y = lds_off;                              // D#.lds_addr (bytes)
  g0.z = (unsigned)ga;                         // global_addr[31:0]
  g0.w = (unsigned)(ga >> 32) | (2u << 30);    // global_addr[56:32] | type=2
  v8i g1;
  g1[0] = (int)((1u << 16)                     // data_size = 2 bytes
              | (1u << 20)                     // pad_enable
              | (pad_int_code << 22)
              | (pad_amt_code << 25));
  g1[1] = (int)((tensor_d0 & 0xFFFFu) << 16);                          // dim0[15:0]
  g1[2] = (int)((tensor_d0 >> 16) | ((tensor_d1 & 0xFFFFu) << 16));    // dim0[31:16]|dim1[15:0]
  g1[3] = (int)((tensor_d1 >> 16) | (tile_d0 << 16));                  // dim1[31:16]|tile_dim0
  g1[4] = (int)tile_d1;                                                // tile_dim1 (tile_dim2=0)
  g1[5] = (int)stride0;                                                // dim0_stride[31:0]
  g1[6] = 0;                                                           // stride0[47:32]|stride1 lo
  g1[7] = 0;
  v4i z4 = {0, 0, 0, 0};
#if defined(__clang_major__) && (__clang_major__ >= 23)
  v8i z8 = {0, 0, 0, 0, 0, 0, 0, 0};
  __builtin_amdgcn_tensor_load_to_lds(g0, g1, z4, z4, z8, 0);
#else
  __builtin_amdgcn_tensor_load_to_lds(g0, g1, z4, z4, 0);
#endif
}

// ---------------- fp32 -> bf16 converts ----------------
__global__ void k_to_bf16(const float* __restrict__ in, unsigned short* __restrict__ out, int n) {
  int i = blockIdx.x * 256 + threadIdx.x;
  if (i < n) out[i] = f2bf(in[i]);
}

// in[k][n] (KxN) -> out[n][k] (NxK) bf16 transpose
__global__ void k_to_bf16_tr(const float* __restrict__ in, unsigned short* __restrict__ out,
                             int K, int N) {
  int i = blockIdx.x * 256 + threadIdx.x;
  if (i < K * N) {
    int k = i / N, n = i - k * N;
    out[(size_t)n * K + k] = f2bf(in[i]);
  }
}

// ---------------- bf16 GEMM: C[MxN] = A[MxK] * Bt[NxK]^T ----------------
// block tile 64(M) x 128(N), 8 waves x (32x32); TDM-fed double-buffered LDS.
__global__ __launch_bounds__(256, 2)
void k_gemm_bf16(const unsigned short* __restrict__ A,
                 const unsigned short* __restrict__ Bt,
                 unsigned short* __restrict__ Cb,     // mode 0: bf16 out
                 float* __restrict__ Cf,              // mode 1: fp32 out
                 const float* __restrict__ resid,     // mode 1: residual add
                 int M, int N, int K, int mode)
{
  constexpr int LDP = 40;  // 32-elem rows + 8-elem pad (TDM: 16 DW interval, 4 DW pad)
  __shared__ __attribute__((aligned(16))) unsigned short sA[2][64  * LDP];
  __shared__ __attribute__((aligned(16))) unsigned short sB[2][128 * LDP];

  const int tid  = threadIdx.x;
  const int lane = tid & 31;
  const int wave = tid >> 5;
  const int wm = wave >> 2;     // 0..1 : 32-row strip
  const int wn = wave & 3;      // 0..3 : 32-col strip
  const int ln = lane & 15;
  const int kh = lane >> 4;

  const int m0 = blockIdx.y * 64;
  const int n0 = blockIdx.x * 128;
  const bool issuer = (wave == 0);

  v8f acc[2][2] = {};

  if (issuer) {
    tdm_load_2d((unsigned)(uintptr_t)&sA[0][0], A  + (size_t)m0 * K,
                (unsigned)K, (unsigned)M, 32, 64, (unsigned)K, 3, 3);
    tdm_load_2d((unsigned)(uintptr_t)&sB[0][0], Bt + (size_t)n0 * K,
                (unsigned)K, (unsigned)N, 32, 128, (unsigned)K, 3, 3);
  }

  for (int kt = 0, it = 0; kt < K; kt += 32, ++it) {
    const int cur = it & 1;
    __builtin_amdgcn_s_wait_tensorcnt(0);   // no-op for non-issuing waves
    __syncthreads();                        // tile `cur` visible; buffer cur^1 free

    if (issuer && (kt + 32 < K)) {          // overlap next DMA with compute
      tdm_load_2d((unsigned)(uintptr_t)&sA[cur ^ 1][0], A  + (size_t)m0 * K + kt + 32,
                  (unsigned)K, (unsigned)M, 32, 64, (unsigned)K, 3, 3);
      tdm_load_2d((unsigned)(uintptr_t)&sB[cur ^ 1][0], Bt + (size_t)n0 * K + kt + 32,
                  (unsigned)K, (unsigned)N, 32, 128, (unsigned)K, 3, 3);
    }

    // A frag (16-bit A layout): lanes0-15 K in {8kh..}+{16+8kh..}
    Frag16 aF[2], bF[2];
    #pragma unroll
    for (int i = 0; i < 2; ++i) {
      int r = wm * 32 + i * 16 + ln;
      aF[i].q[0] = *(const uint4*)&sA[cur][r * LDP + 8 * kh];
      aF[i].q[1] = *(const uint4*)&sA[cur][r * LDP + 16 + 8 * kh];
    }
    // B frag: lanes0-15 K=0-7,16-23, lanes16-31 K=8-15,24-31 (matches A)
    #pragma unroll
    for (int j = 0; j < 2; ++j) {
      int r = wn * 32 + j * 16 + ln;
      bF[j].q[0] = *(const uint4*)&sB[cur][r * LDP + 8 * kh];
      bF[j].q[1] = *(const uint4*)&sB[cur][r * LDP + 16 + 8 * kh];
    }

    #pragma unroll
    for (int i = 0; i < 2; ++i)
      #pragma unroll
      for (int j = 0; j < 2; ++j)
        acc[i][j] = wmma_bf16(aF[i].v, bF[j].v, acc[i][j]);
  }

  // epilogue: D element (r, lane): row = r + 8*kh, col = ln
  #pragma unroll
  for (int i = 0; i < 2; ++i)
    #pragma unroll
    for (int j = 0; j < 2; ++j)
      #pragma unroll
      for (int r = 0; r < 8; ++r) {
        int row = m0 + wm * 32 + i * 16 + r + 8 * kh;
        int col = n0 + wn * 32 + j * 16 + ln;
        size_t idx = (size_t)row * N + col;
        float v = acc[i][j][r];
        if (mode == 0) Cb[idx] = f2bf(v);
        else           Cf[idx] = v + resid[idx];
      }
}

// ---------------- flash attention (causal), 64 q-rows per block ----------------
__global__ __launch_bounds__(128)
void k_attention(const unsigned short* __restrict__ Qb,
                 const unsigned short* __restrict__ Kb,
                 const unsigned short* __restrict__ Vb,
                 unsigned short* __restrict__ Ctx)
{
  constexpr int LP = 72;  // 64-elem rows + 8-elem pad (TDM: 32 DW interval, 4 DW pad)
  __shared__ __attribute__((aligned(16))) unsigned short sQ[64 * LP];
  __shared__ __attribute__((aligned(16))) unsigned short sK[64 * LP];
  __shared__ __attribute__((aligned(16))) unsigned short sV[64 * LP];    // transposed [d][t]
  __shared__ __attribute__((aligned(16))) unsigned short sP[4][16 * LP]; // per-wave P staging

  const int tid  = threadIdx.x;
  const int lane = tid & 31, wave = tid >> 5;
  const int ln = lane & 15, kh = lane >> 4;
  const int b = blockIdx.z, h = blockIdx.y, qb = blockIdx.x;
  const int q0 = qb * 64;
  const bool issuer = (wave == 0);

  // Q tile 64x64 via TDM (row stride DIM elems)
  if (issuer)
    tdm_load_2d((unsigned)(uintptr_t)&sQ[0],
                Qb + ((size_t)(b * SEQ + q0)) * DIM + h * DH,
                DIM, 64, 64, 64, DIM, 4, 3);
  __builtin_amdgcn_s_wait_tensorcnt(0);
  __syncthreads();

  // Q A-fragments for both K-steps (d: 0-31, 32-63), kept in registers
  Frag16 aQ[2];
  #pragma unroll
  for (int ks = 0; ks < 2; ++ks) {
    int r = wave * 16 + ln;
    aQ[ks].q[0] = *(const uint4*)&sQ[r * LP + ks * 32 + 8 * kh];
    aQ[ks].q[1] = *(const uint4*)&sQ[r * LP + ks * 32 + 16 + 8 * kh];
  }

  float mrow[8], lrow[8];
  #pragma unroll
  for (int r = 0; r < 8; ++r) { mrow[r] = -1e30f; lrow[r] = 0.f; }
  v8f o[4] = {};

  for (int kb = 0; kb <= qb; ++kb) {
    // K tile [t][d] via TDM (row-major == transposed-B layout for scores)
    if (issuer)
      tdm_load_2d((unsigned)(uintptr_t)&sK[0],
                  Kb + ((size_t)(b * SEQ + kb * 64)) * DIM + h * DH,
                  DIM, 64, 64, 64, DIM, 4, 3);
    // V tile transposed into sV[d][t] (TDM has no transpose) — manual
    {
      int t  = tid >> 1;
      int d0 = (tid & 1) * 32;
      const unsigned short* src = Vb + ((size_t)(b * SEQ + kb * 64 + t)) * DIM + h * DH + d0;
      #pragma unroll
      for (int c = 0; c < 4; ++c) {
        union { uint4 u; unsigned short s[8]; } w;
        w.u = *(const uint4*)(src + c * 8);
        #pragma unroll
        for (int j = 0; j < 8; ++j)
          sV[(d0 + c * 8 + j) * LP + t] = w.s[j];
      }
    }
    __builtin_amdgcn_s_wait_tensorcnt(0);
    __syncthreads();

    // scores strip [16 x 64] per wave: S = Q K^T
    v8f sa[4] = {};
    #pragma unroll
    for (int ks = 0; ks < 2; ++ks)
      #pragma unroll
      for (int nt = 0; nt < 4; ++nt) {
        Frag16 bK;
        int r = nt * 16 + ln;
        bK.q[0] = *(const uint4*)&sK[r * LP + ks * 32 + 8 * kh];
        bK.q[1] = *(const uint4*)&sK[r * LP + ks * 32 + 16 + 8 * kh];
        sa[nt] = wmma_bf16(aQ[ks].v, bK.v, sa[nt]);
      }

    // scale + causal mask (only diagonal block needs element masking)
    float sc[4][8];
    const bool diag = (kb == qb);
    #pragma unroll
    for (int nt = 0; nt < 4; ++nt)
      #pragma unroll
      for (int r = 0; r < 8; ++r) {
        float s = sa[nt][r] * 0.125f;            // 1/sqrt(64)
        if (diag) {
          int tg = nt * 16 + ln;
          int qg = wave * 16 + r + 8 * kh;
          if (tg > qg) s = -1e30f;
        }
        sc[nt][r] = s;
      }

    // online softmax; each row lives on 16 lanes (n) x 4 tiles
    #pragma unroll
    for (int r = 0; r < 8; ++r) {
      float rm = sc[0][r];
      #pragma unroll
      for (int nt = 1; nt < 4; ++nt) rm = fmaxf(rm, sc[nt][r]);
      #pragma unroll
      for (int off = 1; off < 16; off <<= 1)
        rm = fmaxf(rm, __shfl_xor(rm, off, 32));
      float mn   = fmaxf(mrow[r], rm);
      float corr = __expf(mrow[r] - mn);
      float rs = 0.f;
      #pragma unroll
      for (int nt = 0; nt < 4; ++nt) {
        float p = __expf(sc[nt][r] - mn);
        sc[nt][r] = p;
        rs += p;
      }
      #pragma unroll
      for (int off = 1; off < 16; off <<= 1)
        rs += __shfl_xor(rs, off, 32);
      lrow[r] = lrow[r] * corr + rs;
      mrow[r] = mn;
      #pragma unroll
      for (int nt = 0; nt < 4; ++nt)
        o[nt][r] = o[nt][r] * corr;
    }

    // stage P (bf16) into per-wave LDS region, read back as A-fragments
    #pragma unroll
    for (int nt = 0; nt < 4; ++nt)
      #pragma unroll
      for (int r = 0; r < 8; ++r)
        sP[wave][(r + 8 * kh) * LP + nt * 16 + ln] = f2bf(sc[nt][r]);

    // O += P @ V  (contraction over t, 2 k-steps of 32)
    #pragma unroll
    for (int ks = 0; ks < 2; ++ks) {
      Frag16 aP;
      aP.q[0] = *(const uint4*)&sP[wave][ln * LP + ks * 32 + 8 * kh];
      aP.q[1] = *(const uint4*)&sP[wave][ln * LP + ks * 32 + 16 + 8 * kh];
      #pragma unroll
      for (int nt = 0; nt < 4; ++nt) {
        Frag16 bV;
        int r = nt * 16 + ln;
        bV.q[0] = *(const uint4*)&sV[r * LP + ks * 32 + 8 * kh];
        bV.q[1] = *(const uint4*)&sV[r * LP + ks * 32 + 16 + 8 * kh];
        o[nt] = wmma_bf16(aP.v, bV.v, o[nt]);
      }
    }
    __syncthreads();  // all reads of sK/sV done before next key block's DMA/stores
  }

  // epilogue: ctx = O / l, bf16, row-major [B*S, H*DH]
  #pragma unroll
  for (int nt = 0; nt < 4; ++nt)
    #pragma unroll
    for (int r = 0; r < 8; ++r) {
      float v = o[nt][r] / lrow[r];
      size_t row = (size_t)(b * SEQ + q0 + wave * 16 + r + 8 * kh);
      Ctx[row * DIM + h * DH + nt * 16 + ln] = f2bf(v);
    }
}

// ---------------- in-place LayerNorm over last dim (1024) ----------------
__global__ __launch_bounds__(256)
void k_layernorm(float* __restrict__ out,
                 const float* __restrict__ gamma,
                 const float* __restrict__ beta)
{
  __shared__ float rsum[8], rsq[8];
  const int row = blockIdx.x;
  float* p = out + (size_t)row * DIM;
  const int tid = threadIdx.x;
  float v[4];
  float s = 0.f, s2 = 0.f;
  #pragma unroll
  for (int i = 0; i < 4; ++i) {
    v[i] = p[tid + i * 256];
    s += v[i]; s2 += v[i] * v[i];
  }
  #pragma unroll
  for (int off = 16; off >= 1; off >>= 1) {
    s  += __shfl_xor(s,  off, 32);
    s2 += __shfl_xor(s2, off, 32);
  }
  if ((tid & 31) == 0) { rsum[tid >> 5] = s; rsq[tid >> 5] = s2; }
  __syncthreads();
  float ts = 0.f, ts2 = 0.f;
  #pragma unroll
  for (int w = 0; w < 8; ++w) { ts += rsum[w]; ts2 += rsq[w]; }
  float mean = ts * (1.f / 1024.f);
  float var  = ts2 * (1.f / 1024.f) - mean * mean;
  float inv  = rsqrtf(var + 1e-5f);
  #pragma unroll
  for (int i = 0; i < 4; ++i) {
    int c = tid + i * 256;
    p[c] = (v[i] - mean) * inv * gamma[c] + beta[c];
  }
}

extern "C" void kernel_launch(void* const* d_in, const int* in_sizes, int n_in,
                              void* d_out, int out_size, void* d_ws, size_t ws_size,
                              hipStream_t stream)
{
  (void)in_sizes; (void)n_in; (void)out_size; (void)ws_size;
  const float* x     = (const float*)d_in[0];
  const float* WQ    = (const float*)d_in[1];
  const float* WK    = (const float*)d_in[2];
  const float* WV    = (const float*)d_in[3];
  const float* WO    = (const float*)d_in[4];
  const float* gamma = (const float*)d_in[5];
  const float* beta  = (const float*)d_in[6];

  const size_t M  = (size_t)NB * SEQ;   // 8192 rows
  const size_t MS = M * DIM;            // 8.39M elems
  const size_t WS = (size_t)DIM * DIM;  // 1.05M elems

  unsigned short* p = (unsigned short*)d_ws;
  unsigned short* xb  = p; p += MS;     // x in bf16
  unsigned short* wqT = p; p += WS;     // weights transposed bf16
  unsigned short* wkT = p; p += WS;
  unsigned short* wvT = p; p += WS;
  unsigned short* woT = p; p += WS;
  unsigned short* Qb  = p; p += MS;
  unsigned short* Kb  = p; p += MS;
  unsigned short* Vb  = p; p += MS;
  unsigned short* Cx  = p; p += MS;

  // converts
  k_to_bf16<<<dim3((unsigned)((MS + 255) / 256)), dim3(256), 0, stream>>>(x, xb, (int)MS);
  k_to_bf16_tr<<<dim3((unsigned)((WS + 255) / 256)), dim3(256), 0, stream>>>(WQ, wqT, DIM, DIM);
  k_to_bf16_tr<<<dim3((unsigned)((WS + 255) / 256)), dim3(256), 0, stream>>>(WK, wkT, DIM, DIM);
  k_to_bf16_tr<<<dim3((unsigned)((WS + 255) / 256)), dim3(256), 0, stream>>>(WV, wvT, DIM, DIM);
  k_to_bf16_tr<<<dim3((unsigned)((WS + 255) / 256)), dim3(256), 0, stream>>>(WO, woT, DIM, DIM);

  // QKV projections
  dim3 gg(DIM / 128, (unsigned)(M / 64));
  k_gemm_bf16<<<gg, 256, 0, stream>>>(xb, wqT, Qb, nullptr, nullptr, (int)M, DIM, DIM, 0);
  k_gemm_bf16<<<gg, 256, 0, stream>>>(xb, wkT, Kb, nullptr, nullptr, (int)M, DIM, DIM, 0);
  k_gemm_bf16<<<gg, 256, 0, stream>>>(xb, wvT, Vb, nullptr, nullptr, (int)M, DIM, DIM, 0);

  // causal flash attention
  k_attention<<<dim3(SEQ / 64, NH, NB), dim3(128), 0, stream>>>(Qb, Kb, Vb, Cx);

  // output projection + residual (fp32 out), then in-place LayerNorm
  k_gemm_bf16<<<gg, 256, 0, stream>>>(Cx, woT, nullptr, (float*)d_out, x, (int)M, DIM, DIM, 1);
  k_layernorm<<<dim3((unsigned)M), dim3(256), 0, stream>>>((float*)d_out, gamma, beta);
}
